// EdgeModel_18786186952979
// MI455X (gfx1250) — matrix operand order repaired and checked by reading, not compiled, not run on hardware
//
#include <hip/hip_runtime.h>

// ---------- types ----------
typedef __attribute__((ext_vector_type(16))) __bf16   v16bf;
typedef __attribute__((ext_vector_type(2)))  __bf16   v2bf;
typedef __attribute__((ext_vector_type(8)))  float    v8f;
typedef __attribute__((ext_vector_type(4)))  unsigned uint4v;
typedef __attribute__((ext_vector_type(4)))  int      int4v;
typedef __attribute__((ext_vector_type(2)))  float    float2v;

union FragU {
    unsigned u[8];
    v2bf     h[8];
    uint4v   q[2];
    v16bf    v;
};

#if defined(__has_builtin)
#if __has_builtin(__builtin_amdgcn_cvt_pk_bf16_f32)
#define HAVE_CVT_PK_BF16 1
#endif
#if __has_builtin(__builtin_amdgcn_global_load_async_to_lds_b128)
#define USE_ASYNC_LDS 1
#endif
#endif

__device__ __forceinline__ unsigned pack_bf16_manual(float a, float b) {
    union { float f; unsigned u; } ua, ub;
    ua.f = a; ub.f = b;
    unsigned ra = (ua.u + 0x7FFFu + ((ua.u >> 16) & 1u)) >> 16;   // RNE
    unsigned rb = (ub.u + 0x7FFFu + ((ub.u >> 16) & 1u)) >> 16;
    return (rb << 16) | (ra & 0xFFFFu);
}

__device__ __forceinline__ void set_pair(FragU& f, int v, float a, float b) {
#if HAVE_CVT_PK_BF16
    f.h[v] = __builtin_amdgcn_cvt_pk_bf16_f32(a, b);
#else
    f.u[v] = pack_bf16_manual(a, b);
#endif
}

__device__ __forceinline__ v8f vzero8() {
    v8f z = {0.f, 0.f, 0.f, 0.f, 0.f, 0.f, 0.f, 0.f};
    return z;
}

__device__ __forceinline__ v8f wmma_bf16(const FragU& A, const FragU& B, v8f C) {
    return __builtin_amdgcn_wmma_f32_16x16x32_bf16(
        false, A.v, false, B.v, (short)0, C, false, false);
}

// ---------- LDS layout (dwords) ----------
// W1 fragments: 384*128/2 = 24576 dwords (96 KB bf16)
// W2 fragments: 128*128/2 =  8192 dwords (32 KB)
// W3 fragments:              8192 dwords (32 KB)
// per-wave activation buffer: 32x128 f32 = 4096 dwords, 8 waves (128 KB)
#define LDS_W1     0
#define LDS_W2     24576
#define LDS_W3     32768
#define LDS_H      40960
#define LDS_DWORDS (LDS_H + 8 * 4096)          /* 73728 dwords = 288 KB */
#define SMEM_BYTES (LDS_DWORDS * 4)

// ============================================================
// Prep: repack f32 weights [K,128] into bf16 WMMA B-fragments.
// Fragment (n,t): 32 lanes x 8 dwords. Lane L: N-col = 16n + L%16,
// K = 32t + 16*(L/16) + 2v + p  (p packed in dword halves).
// ============================================================
__global__ void prep_weights_kernel(const float* __restrict__ W,
                                    unsigned* __restrict__ out, int Kdim) {
    const int nKt   = Kdim >> 5;
    const int total = (Kdim * 128) >> 1;
    for (int d = blockIdx.x * blockDim.x + threadIdx.x; d < total;
         d += gridDim.x * blockDim.x) {
        int frag = d >> 8, rem = d & 255;
        int L = rem >> 3, v = rem & 7;
        int n = frag / nKt, t = frag - n * nKt;
        int k   = (t << 5) + ((L >> 4) << 4) + (v << 1);
        int col = (n << 4) + (L & 15);
        float a = W[k * 128 + col];
        float b = W[(k + 1) * 128 + col];
        out[d] = pack_bf16_manual(a, b);
    }
}

// ============================================================
// Edge MLP: each wave owns a 32-edge tile (two M-halves sharing
// every B-fragment -> half the LDS weight traffic per FLOP).
// 320 v_wmma_f32_16x16x32_bf16 per tile; B fragments ping-pong
// double-buffered from LDS; LayerNorm via wave32 xor-shuffles;
// residual add fused into the store.
// ============================================================
__global__ void __launch_bounds__(256)
edge_mlp_kernel(const float* __restrict__ nf,      // [N,128] node features
                const int*   __restrict__ snd,     // [E]
                const int*   __restrict__ rcv,     // [E]
                const float* __restrict__ ef,      // [E,128] edge features
                const unsigned* __restrict__ w1f,  // bf16 fragments
                const unsigned* __restrict__ w2f,
                const unsigned* __restrict__ w3f,
                const float* __restrict__ b1,
                const float* __restrict__ b2,
                const float* __restrict__ b3,
                const float* __restrict__ lng,
                const float* __restrict__ lnb,
                float* __restrict__ out,           // [E,128]
                int nEdges) {
    extern __shared__ unsigned smem[];

    const int lane = threadIdx.x & 31;
    const int wave = threadIdx.x >> 5;
    const int m    = lane & 15;     // A-matrix row-in-half / C-matrix column
    const int half = lane >> 4;     // lane group (selects K / M halves)

    // ---- stage all weights into LDS once per block ----
#if USE_ASYNC_LDS
    for (int i = threadIdx.x * 4; i < 24576; i += 256 * 4)
        __builtin_amdgcn_global_load_async_to_lds_b128(
            (__attribute__((address_space(1))) int4v*)(w1f + i),
            (__attribute__((address_space(3))) int4v*)(&smem[LDS_W1 + i]), 0, 0);
    for (int i = threadIdx.x * 4; i < 8192; i += 256 * 4) {
        __builtin_amdgcn_global_load_async_to_lds_b128(
            (__attribute__((address_space(1))) int4v*)(w2f + i),
            (__attribute__((address_space(3))) int4v*)(&smem[LDS_W2 + i]), 0, 0);
        __builtin_amdgcn_global_load_async_to_lds_b128(
            (__attribute__((address_space(1))) int4v*)(w3f + i),
            (__attribute__((address_space(3))) int4v*)(&smem[LDS_W3 + i]), 0, 0);
    }
#if __has_builtin(__builtin_amdgcn_s_wait_asynccnt)
    __builtin_amdgcn_s_wait_asynccnt(0);
#else
    asm volatile("s_wait_asynccnt 0x0" ::: "memory");
#endif
#else
    for (int i = threadIdx.x; i < 24576; i += 256) smem[LDS_W1 + i] = w1f[i];
    for (int i = threadIdx.x; i < 8192; i += 256) {
        smem[LDS_W2 + i] = w2f[i];
        smem[LDS_W3 + i] = w3f[i];
    }
#endif
    __syncthreads();

    float* hb = reinterpret_cast<float*>(&smem[LDS_H]) + wave * 4096;

    auto loadB = [&](int base, int fragIdx) {
        FragU b;
        const uint4v* bp = reinterpret_cast<const uint4v*>(
            &smem[base + fragIdx * 256 + lane * 8]);
        b.q[0] = bp[0];
        b.q[1] = bp[1];
        return b;
    };

    const int nTiles = (nEdges + 31) >> 5;
    const int tStride = gridDim.x * 8;

    for (int tile = blockIdx.x * 8 + wave; tile < nTiles; tile += tStride) {
        const int e0 = tile << 5;

        // prefetch next tile's gather metadata / edge rows
        const int nxt = tile + tStride;
        if (nxt < nTiles) {
            __builtin_prefetch(&snd[nxt << 5], 0, 1);
            __builtin_prefetch(&rcv[nxt << 5], 0, 1);
            __builtin_prefetch(&ef[(size_t)(nxt << 5) * 128], 0, 1);
        }

        int ea = e0 + m;         if (ea >= nEdges) ea = nEdges - 1;
        int eb = e0 + 16 + m;    if (eb >= nEdges) eb = nEdges - 1;

        const float* ps0 = nf + (size_t)snd[ea] * 128;
        const float* pr0 = nf + (size_t)rcv[ea] * 128;
        const float* pe0 = ef + (size_t)ea * 128;
        const float* ps1 = nf + (size_t)snd[eb] * 128;
        const float* pr1 = nf + (size_t)rcv[eb] * 128;
        const float* pe1 = ef + (size_t)eb * 128;

        v8f acc0[8], acc1[8];
        #pragma unroll
        for (int n = 0; n < 8; ++n) { acc0[n] = vzero8(); acc1[n] = vzero8(); }

        // ------------- Layer 1: [32x384] @ [384x128] -------------
        auto l1seg = [&](const float* s0, const float* s1, int tbase) {
            for (int t = 0; t < 4; ++t) {
                FragU a0, a1;
                #pragma unroll
                for (int v = 0; v < 8; ++v) {
                    int k = (t << 5) + (half << 4) + (v << 1);
                    float2v f0 = *reinterpret_cast<const float2v*>(s0 + k);
                    float2v f1 = *reinterpret_cast<const float2v*>(s1 + k);
                    set_pair(a0, v, f0.x, f0.y);
                    set_pair(a1, v, f1.x, f1.y);
                }
                const int T = tbase + t;
                FragU b = loadB(LDS_W1, T);          // frag (n=0, T)
                #pragma unroll
                for (int n = 0; n < 8; ++n) {
                    FragU bn;
                    if (n < 7) bn = loadB(LDS_W1, (n + 1) * 12 + T);
                    acc0[n] = wmma_bf16(a0, b, acc0[n]);
                    acc1[n] = wmma_bf16(a1, b, acc1[n]);
                    if (n < 7) b = bn;
                }
            }
        };
        l1seg(ps0, ps1, 0);   // sender node features   (k 0..127)
        l1seg(pr0, pr1, 4);   // receiver node features (k 128..255)
        l1seg(pe0, pe1, 8);   // edge features          (k 256..383)

        // bias + relu -> per-wave LDS activation buffer (32x128 f32)
        #pragma unroll
        for (int n = 0; n < 8; ++n) {
            const float bias = b1[(n << 4) + m];
            #pragma unroll
            for (int r = 0; r < 8; ++r) {
                float x0 = acc0[n][r] + bias; x0 = x0 > 0.f ? x0 : 0.f;
                float x1 = acc1[n][r] + bias; x1 = x1 > 0.f ? x1 : 0.f;
                const int c = (n << 4) + m;
                hb[(r + (half << 3)) * 128 + c]      = x0;
                hb[(16 + r + (half << 3)) * 128 + c] = x1;
            }
        }

        // ------------- hidden layers: [32x128] @ [128x128] -------------
        auto hidden = [&](int wbase, const float* bias, bool relu, bool store) {
            #pragma unroll
            for (int n = 0; n < 8; ++n) { acc0[n] = vzero8(); acc1[n] = vzero8(); }
            for (int t = 0; t < 4; ++t) {
                FragU a0, a1;
                #pragma unroll
                for (int v = 0; v < 8; ++v) {
                    int k = (t << 5) + (half << 4) + (v << 1);
                    float2v f0 = *reinterpret_cast<const float2v*>(hb + m * 128 + k);
                    float2v f1 = *reinterpret_cast<const float2v*>(hb + (m + 16) * 128 + k);
                    set_pair(a0, v, f0.x, f0.y);
                    set_pair(a1, v, f1.x, f1.y);
                }
                FragU b = loadB(wbase, t);           // frag (n=0, t)
                #pragma unroll
                for (int n = 0; n < 8; ++n) {
                    FragU bn;
                    if (n < 7) bn = loadB(wbase, (n + 1) * 4 + t);
                    acc0[n] = wmma_bf16(a0, b, acc0[n]);
                    acc1[n] = wmma_bf16(a1, b, acc1[n]);
                    if (n < 7) b = bn;
                }
            }
            #pragma unroll
            for (int n = 0; n < 8; ++n) {
                const float bv = bias[(n << 4) + m];
                #pragma unroll
                for (int r = 0; r < 8; ++r) {
                    float x0 = acc0[n][r] + bv;
                    float x1 = acc1[n][r] + bv;
                    if (relu) { x0 = x0 > 0.f ? x0 : 0.f; x1 = x1 > 0.f ? x1 : 0.f; }
                    acc0[n][r] = x0;
                    acc1[n][r] = x1;
                    if (store) {
                        const int c = (n << 4) + m;
                        hb[(r + (half << 3)) * 128 + c]      = x0;
                        hb[(16 + r + (half << 3)) * 128 + c] = x1;
                    }
                }
            }
        };
        hidden(LDS_W2, b2, true,  true);    // layer 2
        hidden(LDS_W3, b3, false, false);   // layer 3 (stays in acc)

        // ------------- LayerNorm + affine + residual + store -------------
        // C-layout: lane holds column 16n+m of rows (16*hh + r + 8*half).
        // xor masks 1,2,4,8 reduce over the 16 lanes of each half-group.
        #pragma unroll
        for (int hh = 0; hh < 2; ++hh) {
            v8f* acc = hh ? acc1 : acc0;
            float mean[8], rstd[8];
            #pragma unroll
            for (int r = 0; r < 8; ++r) {
                float s = 0.f, q = 0.f;
                #pragma unroll
                for (int n = 0; n < 8; ++n) {
                    float x = acc[n][r];
                    s += x;
                    q += x * x;
                }
                s += __shfl_xor(s, 1); q += __shfl_xor(q, 1);
                s += __shfl_xor(s, 2); q += __shfl_xor(q, 2);
                s += __shfl_xor(s, 4); q += __shfl_xor(q, 4);
                s += __shfl_xor(s, 8); q += __shfl_xor(q, 8);
                float mu  = s * (1.f / 128.f);
                float var = q * (1.f / 128.f) - mu * mu;
                mean[r] = mu;
                rstd[r] = rsqrtf(var + 1e-5f);
            }
            #pragma unroll
            for (int n = 0; n < 8; ++n) {
                const int c = (n << 4) + m;
                const float gv = lng[c], bv = lnb[c];
                #pragma unroll
                for (int r = 0; r < 8; ++r) {
                    const int row = e0 + (hh << 4) + r + (half << 3);
                    if (row < nEdges) {
                        const size_t off = (size_t)row * 128 + c;
                        out[off] = (acc[n][r] - mean[r]) * rstd[r] * gv + bv + ef[off];
                    }
                }
            }
        }
    }
}

// ============================================================
// Host launcher
// ============================================================
static inline int imin(int a, int b) { return a < b ? a : b; }

extern "C" void kernel_launch(void* const* d_in, const int* in_sizes, int n_in,
                              void* d_out, int out_size, void* d_ws, size_t ws_size,
                              hipStream_t stream) {
    const float* nf   = (const float*)d_in[0];
    const int*   msnd = (const int*)d_in[1];
    const int*   mrcv = (const int*)d_in[2];
    const float* mef  = (const float*)d_in[3];
    const int*   wsnd = (const int*)d_in[4];
    const int*   wrcv = (const int*)d_in[5];
    const float* wef  = (const float*)d_in[6];
    // mesh params: W1 b1 W2 b2 W3 b3 ln_g ln_b
    const float* mW1 = (const float*)d_in[7];
    const float* mb1 = (const float*)d_in[8];
    const float* mW2 = (const float*)d_in[9];
    const float* mb2 = (const float*)d_in[10];
    const float* mW3 = (const float*)d_in[11];
    const float* mb3 = (const float*)d_in[12];
    const float* mlg = (const float*)d_in[13];
    const float* mlb = (const float*)d_in[14];
    // world params
    const float* wW1 = (const float*)d_in[15];
    const float* wb1 = (const float*)d_in[16];
    const float* wW2 = (const float*)d_in[17];
    const float* wb2 = (const float*)d_in[18];
    const float* wW3 = (const float*)d_in[19];
    const float* wb3 = (const float*)d_in[20];
    const float* wlg = (const float*)d_in[21];
    const float* wlb = (const float*)d_in[22];

    const int Em = in_sizes[1];                       // 200000
    const int Ew = in_sizes[4];                       // 100000
    const size_t meshOutElems = (size_t)in_sizes[3];  // Em*128

    // workspace: bf16 weight fragments (dwords)
    unsigned* ws   = (unsigned*)d_ws;
    unsigned* fmW1 = ws;                 // 24576
    unsigned* fmW2 = ws + 24576;         //  8192
    unsigned* fmW3 = ws + 32768;         //  8192
    unsigned* fwW1 = ws + 40960;         // 24576
    unsigned* fwW2 = ws + 65536;         //  8192
    unsigned* fwW3 = ws + 73728;         //  8192

    prep_weights_kernel<<<48, 256, 0, stream>>>(mW1, fmW1, 384);
    prep_weights_kernel<<<16, 256, 0, stream>>>(mW2, fmW2, 128);
    prep_weights_kernel<<<16, 256, 0, stream>>>(mW3, fmW3, 128);
    prep_weights_kernel<<<48, 256, 0, stream>>>(wW1, fwW1, 384);
    prep_weights_kernel<<<16, 256, 0, stream>>>(wW2, fwW2, 128);
    prep_weights_kernel<<<16, 256, 0, stream>>>(wW3, fwW3, 128);

    (void)hipFuncSetAttribute(reinterpret_cast<const void*>(&edge_mlp_kernel),
                              hipFuncAttributeMaxDynamicSharedMemorySize,
                              SMEM_BYTES);

    float* outMesh  = (float*)d_out;
    float* outWorld = (float*)d_out + meshOutElems;

    const int tilesM = (Em + 31) >> 5;   // 6250
    const int tilesW = (Ew + 31) >> 5;   // 3125
    const int blkM = imin((tilesM + 7) >> 3, 640);
    const int blkW = imin((tilesW + 7) >> 3, 640);

    edge_mlp_kernel<<<blkM, 256, SMEM_BYTES, stream>>>(
        nf, msnd, mrcv, mef, fmW1, fmW2, fmW3,
        mb1, mb2, mb3, mlg, mlb, outMesh, Em);

    edge_mlp_kernel<<<blkW, 256, SMEM_BYTES, stream>>>(
        nf, wsnd, wrcv, wef, fwW1, fwW2, fwW3,
        wb1, wb2, wb3, wlg, wlb, outWorld, Ew);
}
